// RetentiveNetwork_19284403159610
// MI455X (gfx1250) — matrix operand order repaired and checked by reading, not compiled, not run on hardware
//
#include <hip/hip_runtime.h>
#include <cstddef>

// ---------------- problem constants ----------------
#define LAYERS 2
#define HDIM   1024
#define NHEAD  8
#define FFDIM  4096
#define HEADD  128
#define BBAT   2
#define SEQ    2048
#define MTOK   (BBAT*SEQ)   // 4096 tokens

typedef _Float16 h16;
typedef _Float16 v16h __attribute__((ext_vector_type(16)));
typedef float    v8f  __attribute__((ext_vector_type(8)));
typedef unsigned int v4u __attribute__((ext_vector_type(4)));
typedef int          v4i __attribute__((ext_vector_type(4)));
typedef int          v8i __attribute__((ext_vector_type(8)));

union Frag { v16h v; uint4 u[2]; };

#if __has_builtin(__builtin_amdgcn_tensor_load_to_lds) && \
    __has_builtin(__builtin_amdgcn_s_wait_tensorcnt)
#define USE_TDM 1
#else
#define USE_TDM 0
#endif

__device__ __forceinline__ v8f wmma_f16(v16h a, v16h b, v8f c) {
  // D = A(16x32) * B(32x16) + C, f32 accumulate (gfx1250)
  return __builtin_amdgcn_wmma_f32_16x16x32_f16(false, a, false, b, (short)0, c,
                                                false, false);
}

#if USE_TDM
// Issue a TDM 2D tile load: global (row-major, rowstride elems of h16) -> LDS,
// packed [tile_d1][tile_d0] h16. Tracked on TENSORcnt.
__device__ __forceinline__ void tdm_load_2d(unsigned lds_off, const void* gaddr,
                                            unsigned tensor_d0,
                                            unsigned tensor_d1,
                                            unsigned long long rowstride,
                                            unsigned tile_d0, unsigned tile_d1) {
  unsigned long long ga = (unsigned long long)(size_t)gaddr;
  v4u g0;
  g0[0] = 1u;                                   // count=1, user descriptor
  g0[1] = lds_off;                              // LDS byte address
  g0[2] = (unsigned)(ga & 0xFFFFFFFFu);         // global_addr[31:0]
  g0[3] = (unsigned)((ga >> 32) & 0x01FFFFFFu)  // global_addr[56:32]
          | (2u << 30);                         // type=2 ("image")
  v8i g1;
  g1[0] = (int)(1u << 16);                      // data_size=1 (2 bytes)
  g1[1] = (int)((tensor_d0 & 0xFFFFu) << 16);   // tensor_dim0[15:0]
  g1[2] = (int)((tensor_d0 >> 16) | ((tensor_d1 & 0xFFFFu) << 16));
  g1[3] = (int)((tensor_d1 >> 16) | (tile_d0 << 16));  // tile_dim0
  g1[4] = (int)(tile_d1 & 0xFFFFu);             // tile_dim1 (tile_dim2=0)
  g1[5] = (int)(unsigned)(rowstride & 0xFFFFFFFFull);  // dim0_stride[31:0]
  g1[6] = (int)(unsigned)((rowstride >> 32) & 0xFFFFull);  // [47:32], s1=0
  g1[7] = 0;
  v4i g2 = {};  // dims 2/3 unused
  v4i g3 = {};
#if defined(__clang_major__) && (__clang_major__ >= 23)
  v8i g4 = {};
  __builtin_amdgcn_tensor_load_to_lds(g0, g1, g2, g3, g4, 0);
#else
  __builtin_amdgcn_tensor_load_to_lds(g0, g1, g2, g3, 0);
#endif
}
#endif  // USE_TDM

// ---------------- weight convert + transpose (f32[R,C] -> f16[C,R]) ----------
__global__ __launch_bounds__(256)
void cvt_transpose_k(const float* __restrict__ src, h16* __restrict__ dst,
                     int R, int C) {
  int idx = blockIdx.x * 256 + threadIdx.x;
  if (idx < R * C) {
    int r = idx / C, c = idx - r * C;
    dst[(size_t)c * R + r] = (h16)src[idx];
  }
}

// ---------------- layernorm: f32 row -> f16 row (H=1024, 256 thr) -----------
__global__ __launch_bounds__(256)
void layernorm_k(const float* __restrict__ x, h16* __restrict__ out,
                 const float* __restrict__ g, const float* __restrict__ b) {
  const int row = blockIdx.x;
  const int tid = threadIdx.x;
  const float* xr = x + (size_t)row * HDIM + tid * 4;
  float v0 = xr[0], v1 = xr[1], v2 = xr[2], v3 = xr[3];
  float s = v0 + v1 + v2 + v3;
  float s2 = v0 * v0 + v1 * v1 + v2 * v2 + v3 * v3;
  for (int o = 16; o; o >>= 1) {
    s += __shfl_xor(s, o, 32);
    s2 += __shfl_xor(s2, o, 32);
  }
  __shared__ float rs[8], rq[8];
  if ((tid & 31) == 0) { rs[tid >> 5] = s; rq[tid >> 5] = s2; }
  __syncthreads();
  s = 0.f; s2 = 0.f;
  for (int i = 0; i < 8; ++i) { s += rs[i]; s2 += rq[i]; }
  float mu = s * (1.f / HDIM);
  float var = s2 * (1.f / HDIM) - mu * mu;
  float inv = rsqrtf(var + 1e-5f);
  const float* gp = g + tid * 4;
  const float* bp = b + tid * 4;
  h16* op = out + (size_t)row * HDIM + tid * 4;
  op[0] = (h16)((v0 - mu) * inv * gp[0] + bp[0]);
  op[1] = (h16)((v1 - mu) * inv * gp[1] + bp[1]);
  op[2] = (h16)((v2 - mu) * inv * gp[2] + bp[2]);
  op[3] = (h16)((v3 - mu) * inv * gp[3] + bp[3]);
}

// ---------------- per-head group norm + swish gate ---------------------------
__global__ __launch_bounds__(256)
void gnorm_gate_k(const float* __restrict__ o, const float* __restrict__ gpre,
                  h16* __restrict__ out) {
  const int row = blockIdx.x;        // b*SEQ + s
  const int hv = threadIdx.x >> 5;   // head (8 waves)
  const int ln = threadIdx.x & 31;
  const float* op = o + (size_t)row * HDIM + hv * HEADD + ln * 4;
  float v0 = op[0], v1 = op[1], v2 = op[2], v3 = op[3];
  float s = v0 + v1 + v2 + v3;
  float s2 = v0 * v0 + v1 * v1 + v2 * v2 + v3 * v3;
  for (int off = 16; off; off >>= 1) {
    s += __shfl_xor(s, off, 32);
    s2 += __shfl_xor(s2, off, 32);
  }
  float mu = s * (1.f / HEADD);
  float var = s2 * (1.f / HEADD) - mu * mu;
  float inv = rsqrtf(var + 1e-5f);
  const float* gp = gpre + (size_t)row * HDIM + hv * HEADD + ln * 4;
  h16* dp = out + (size_t)row * HDIM + hv * HEADD + ln * 4;
  float vv[4] = {v0, v1, v2, v3};
  for (int i = 0; i < 4; ++i) {
    float gv = gp[i];
    float sw = gv / (1.f + __expf(-gv));
    dp[i] = (h16)(sw * (vv[i] - mu) * inv);
  }
}

// ---------------- WMMA GEMM: C[M,N] = A[M,K] * Bt[N,K]^T ---------------------
// MODE 0: f16 out            MODE 1: f32 out
// MODE 2: f16 out, stored transposed as [B, N, Srows]
// MODE 3: f32 out + resid    MODE 4: f16 out, bias+gelu
// MODE 5: f32 out, bias+resid
template <int MODE>
__global__ __launch_bounds__(256)
void gemm_f16_k(const h16* __restrict__ A, const h16* __restrict__ Bt,
                void* __restrict__ Out, const float* __restrict__ bias,
                const float* __restrict__ resid, int M, int N, int K,
                int Srows) {
  const int tid = threadIdx.x;
  const int lane = tid & 31;
  const int wave = tid >> 5;
  const int wr = wave >> 1;  // 0..3: 32-row group
  const int wc = wave & 1;   // 0..1: 64-col group
  const int bx = blockIdx.x; // N tile
  const int by = blockIdx.y; // M tile
  const int mrow = lane & 15;
  const int hi = (lane < 16) ? 0 : 1;
  const int kbA = hi ? 8 : 0;
  const int kbB = hi ? 8 : 0;

  v8f acc[2][4] = {};

#if USE_TDM
  // double-buffered tiles, filled by the Tensor Data Mover
  __shared__ __align__(16) h16 sA[2][128 * 32];
  __shared__ __align__(16) h16 sB[2][128 * 32];
  const h16* Abase = A + (size_t)(by * 128) * K;
  const h16* Bbase = Bt + (size_t)(bx * 128) * K;
  unsigned laA[2] = {(unsigned)(size_t)(void*)&sA[0][0],
                     (unsigned)(size_t)(void*)&sA[1][0]};
  unsigned laB[2] = {(unsigned)(size_t)(void*)&sB[0][0],
                     (unsigned)(size_t)(void*)&sB[1][0]};
  if (wave == 0) {
    tdm_load_2d(laA[0], Abase, (unsigned)K, 128u, (unsigned long long)K, 32u, 128u);
    tdm_load_2d(laB[0], Bbase, (unsigned)K, 128u, (unsigned long long)K, 32u, 128u);
  }
  int buf = 0;
#else
  __shared__ __align__(16) h16 sA1[128 * 32];
  __shared__ __align__(16) h16 sB1[128 * 32];
#endif

  for (int k0 = 0; k0 < K; k0 += 32) {
#if USE_TDM
    if (wave == 0) {
      if (k0 + 32 < K) {
        // prefetch next K-slice into the other buffer while we compute
        tdm_load_2d(laA[buf ^ 1], Abase + k0 + 32, (unsigned)K, 128u,
                    (unsigned long long)K, 32u, 128u);
        tdm_load_2d(laB[buf ^ 1], Bbase + k0 + 32, (unsigned)K, 128u,
                    (unsigned long long)K, 32u, 128u);
        __builtin_amdgcn_s_wait_tensorcnt(2);  // current pair landed
      } else {
        __builtin_amdgcn_s_wait_tensorcnt(0);
      }
    }
    __syncthreads();
    const h16* pA = &sA[buf][0];
    const h16* pB = &sB[buf][0];
#else
    for (int i = tid; i < 512; i += 256) {
      int r = i >> 2, c = (i & 3) << 3;
      *(uint4*)&sA1[r * 32 + c] =
          *(const uint4*)&A[(size_t)(by * 128 + r) * K + k0 + c];
      *(uint4*)&sB1[r * 32 + c] =
          *(const uint4*)&Bt[(size_t)(bx * 128 + r) * K + k0 + c];
    }
    if (k0 + 32 < K) {
      __builtin_prefetch(&A[(size_t)(by * 128 + (tid >> 1)) * K + k0 + 32], 0, 1);
      __builtin_prefetch(&Bt[(size_t)(bx * 128 + (tid >> 1)) * K + k0 + 32], 0, 1);
    }
    __syncthreads();
    const h16* pA = sA1;
    const h16* pB = sB1;
#endif

    v16h af[2], bf[4];
    for (int i = 0; i < 2; ++i) {
      int r = wr * 32 + i * 16 + mrow;
      Frag f;
      f.u[0] = *(const uint4*)&pA[r * 32 + kbA];
      f.u[1] = *(const uint4*)&pA[r * 32 + kbA + 16];
      af[i] = f.v;
    }
    for (int j = 0; j < 4; ++j) {
      int c = wc * 64 + j * 16 + mrow;
      Frag f;
      f.u[0] = *(const uint4*)&pB[c * 32 + kbB];
      f.u[1] = *(const uint4*)&pB[c * 32 + kbB + 16];
      bf[j] = f.v;
    }
    for (int i = 0; i < 2; ++i)
      for (int j = 0; j < 4; ++j)
        acc[i][j] = wmma_f16(af[i], bf[j], acc[i][j]);
    __syncthreads();
#if USE_TDM
    buf ^= 1;
#endif
  }

  for (int i = 0; i < 2; ++i)
    for (int j = 0; j < 4; ++j)
      for (int r = 0; r < 8; ++r) {
        int row = by * 128 + wr * 32 + i * 16 + r + hi * 8;
        int col = bx * 128 + wc * 64 + j * 16 + mrow;
        float v = acc[i][j][r];
        if (MODE == 0) {
          ((h16*)Out)[(size_t)row * N + col] = (h16)v;
        } else if (MODE == 1) {
          ((float*)Out)[(size_t)row * N + col] = v;
        } else if (MODE == 2) {
          int b = row / Srows, s = row - b * Srows;
          ((h16*)Out)[((size_t)b * N + col) * Srows + s] = (h16)v;
        } else if (MODE == 3) {
          ((float*)Out)[(size_t)row * N + col] =
              v + resid[(size_t)row * N + col];
        } else if (MODE == 4) {
          float x = v + bias[col];
          float t = 0.7978845608028654f * (x + 0.044715f * x * x * x);
          ((h16*)Out)[(size_t)row * N + col] = (h16)(0.5f * x * (1.f + tanhf(t)));
        } else {
          ((float*)Out)[(size_t)row * N + col] =
              v + bias[col] + resid[(size_t)row * N + col];
        }
      }
}

// ---------------- retention: decay-masked attention, WMMA --------------------
// grid (SEQ/16, BBAT*NHEAD), 32 threads (one wave). q,k: [B,S,H] f16.
// vT: [B,H,S] f16 (key index contiguous). o: [B,S,H] f32.
__global__ __launch_bounds__(32)
void retention_k(const h16* __restrict__ q, const h16* __restrict__ k,
                 const h16* __restrict__ vT, float* __restrict__ o) {
  const int n0 = blockIdx.x * 16;
  const int bh = blockIdx.y;
  const int b = bh >> 3;
  const int h = bh & 7;
  const int lane = threadIdx.x;
  const int mrow = lane & 15;
  const int hi = (lane < 16) ? 0 : 1;
  const int kbq = hi ? 8 : 0;
  const int kbB = hi ? 8 : 0;

  __shared__ __align__(16) h16 sS[16 * 32];

  // persistent q fragments: 16 queries x 128 features = 4 A-operands
  v16h qf[4];
  for (int kk = 0; kk < 4; ++kk) {
    const h16* p = q + ((size_t)(b * SEQ + n0 + mrow)) * HDIM + h * HEADD + kk * 32;
    Frag f;
    f.u[0] = *(const uint4*)&p[kbq];
    f.u[1] = *(const uint4*)&p[kbq + 16];
    qf[kk] = f.v;
  }
  const float lgam = logf(1.f - exp2f(-5.f - (float)h));
  const float scale = 0.08838834764831845f;  // HD^-0.5
  v8f oacc[8] = {};

  for (int m0 = 0; m0 <= n0; m0 += 32) {
    // scores: 16x32 block = q(16x128) @ k^T(128x32), decay-masked
    for (int j = 0; j < 2; ++j) {
      v8f s = {};
      for (int kk = 0; kk < 4; ++kk) {
        const h16* p = k + ((size_t)(b * SEQ + m0 + j * 16 + mrow)) * HDIM +
                       h * HEADD + kk * 32 + kbB;
        Frag f;
        f.u[0] = *(const uint4*)p;
        f.u[1] = *(const uint4*)(p + 16);
        s = wmma_f16(qf[kk], f.v, s);
      }
      for (int r = 0; r < 8; ++r) {
        int qn = n0 + r + hi * 8;
        int m = m0 + j * 16 + mrow;
        float d = (m <= qn) ? __expf(lgam * (float)(qn - m)) : 0.f;
        sS[(r + hi * 8) * 32 + j * 16 + mrow] = (h16)(s[r] * scale * d);
      }
    }
    __syncthreads();
    // re-layout scores D->A through LDS, then o(16x128) += S(16x32) @ V(32x128)
    v16h sf;
    {
      Frag f;
      f.u[0] = *(const uint4*)&sS[mrow * 32 + kbq];
      f.u[1] = *(const uint4*)&sS[mrow * 32 + kbq + 16];
      sf = f.v;
    }
    for (int c = 0; c < 8; ++c) {
      const h16* p = vT + ((size_t)(b * HDIM + h * HEADD + c * 16 + mrow)) * SEQ +
                     m0 + kbB;
      Frag f;
      f.u[0] = *(const uint4*)p;
      f.u[1] = *(const uint4*)(p + 16);
      oacc[c] = wmma_f16(sf, f.v, oacc[c]);
    }
    __syncthreads();
  }
  for (int c = 0; c < 8; ++c)
    for (int r = 0; r < 8; ++r)
      o[((size_t)(b * SEQ + n0 + r + hi * 8)) * HDIM + h * HEADD + c * 16 + mrow] =
          oacc[c][r];
}

// ---------------- host orchestration ----------------------------------------
extern "C" void kernel_launch(void* const* d_in, const int* in_sizes, int n_in,
                              void* d_out, int out_size, void* d_ws,
                              size_t ws_size, hipStream_t stream) {
  const float* x_in = (const float*)d_in[0];
  const float* Wq = (const float*)d_in[1];
  const float* Wk = (const float*)d_in[2];
  const float* Wv = (const float*)d_in[3];
  const float* Wg = (const float*)d_in[4];
  const float* Wo = (const float*)d_in[5];
  const float* W1 = (const float*)d_in[6];
  const float* b1 = (const float*)d_in[7];
  const float* W2 = (const float*)d_in[8];
  const float* b2 = (const float*)d_in[9];
  const float* ln_g = (const float*)d_in[10];
  const float* ln_b = (const float*)d_in[11];

  char* base = (char*)d_ws;
  size_t off = 0;
  auto carve = [&](size_t elems, size_t esz) -> char* {
    char* p = base + off;
    off += ((elems * esz) + 255) & ~(size_t)255;
    return p;
  };
  const size_t HH = (size_t)HDIM * HDIM;
  h16* WqT = (h16*)carve((size_t)LAYERS * HH, 2);
  h16* WkT = (h16*)carve((size_t)LAYERS * HH, 2);
  h16* WvT = (h16*)carve((size_t)LAYERS * HH, 2);
  h16* WgT = (h16*)carve((size_t)LAYERS * HH, 2);
  h16* WoT = (h16*)carve((size_t)LAYERS * HH, 2);
  h16* W1T = (h16*)carve((size_t)LAYERS * HDIM * FFDIM, 2);
  h16* W2T = (h16*)carve((size_t)LAYERS * FFDIM * HDIM, 2);
  h16* xn = (h16*)carve((size_t)MTOK * HDIM, 2);
  h16* qb = (h16*)carve((size_t)MTOK * HDIM, 2);
  h16* kb = (h16*)carve((size_t)MTOK * HDIM, 2);
  h16* vT = (h16*)carve((size_t)MTOK * HDIM, 2);
  h16* gob = (h16*)carve((size_t)MTOK * HDIM, 2);
  h16* ynb = (h16*)carve((size_t)MTOK * HDIM, 2);
  h16* hb = (h16*)carve((size_t)MTOK * FFDIM, 2);
  float* gf = (float*)carve((size_t)MTOK * HDIM, 4);
  float* of = (float*)carve((size_t)MTOK * HDIM, 4);
  float* yf = (float*)carve((size_t)MTOK * HDIM, 4);
  float* xf = (float*)carve((size_t)MTOK * HDIM, 4);

  dim3 blk(256);
  // weight conversion (deterministic, every call)
  int gHH = (int)((HH + 255) / 256);
  int gHF = (int)(((size_t)HDIM * FFDIM + 255) / 256);
  for (int l = 0; l < LAYERS; ++l) {
    cvt_transpose_k<<<gHH, blk, 0, stream>>>(Wq + l * HH, WqT + l * HH, HDIM, HDIM);
    cvt_transpose_k<<<gHH, blk, 0, stream>>>(Wk + l * HH, WkT + l * HH, HDIM, HDIM);
    cvt_transpose_k<<<gHH, blk, 0, stream>>>(Wv + l * HH, WvT + l * HH, HDIM, HDIM);
    cvt_transpose_k<<<gHH, blk, 0, stream>>>(Wg + l * HH, WgT + l * HH, HDIM, HDIM);
    cvt_transpose_k<<<gHH, blk, 0, stream>>>(Wo + l * HH, WoT + l * HH, HDIM, HDIM);
    cvt_transpose_k<<<gHF, blk, 0, stream>>>(W1 + (size_t)l * HDIM * FFDIM,
                                             W1T + (size_t)l * FFDIM * HDIM,
                                             HDIM, FFDIM);
    cvt_transpose_k<<<gHF, blk, 0, stream>>>(W2 + (size_t)l * FFDIM * HDIM,
                                             W2T + (size_t)l * HDIM * FFDIM,
                                             FFDIM, HDIM);
  }

  dim3 gridP(HDIM / 128, MTOK / 128);   // 8 x 32 (proj / W2)
  dim3 gridF(FFDIM / 128, MTOK / 128);  // 32 x 32 (W1)
  const float* curx = x_in;
  for (int l = 0; l < LAYERS; ++l) {
    layernorm_k<<<MTOK, blk, 0, stream>>>(curx, xn, ln_g, ln_b);
    gemm_f16_k<0><<<gridP, blk, 0, stream>>>(xn, WqT + l * HH, qb, nullptr,
                                             nullptr, MTOK, HDIM, HDIM, SEQ);
    gemm_f16_k<0><<<gridP, blk, 0, stream>>>(xn, WkT + l * HH, kb, nullptr,
                                             nullptr, MTOK, HDIM, HDIM, SEQ);
    gemm_f16_k<2><<<gridP, blk, 0, stream>>>(xn, WvT + l * HH, vT, nullptr,
                                             nullptr, MTOK, HDIM, HDIM, SEQ);
    gemm_f16_k<1><<<gridP, blk, 0, stream>>>(xn, WgT + l * HH, gf, nullptr,
                                             nullptr, MTOK, HDIM, HDIM, SEQ);
    retention_k<<<dim3(SEQ / 16, BBAT * NHEAD), dim3(32), 0, stream>>>(qb, kb,
                                                                       vT, of);
    gnorm_gate_k<<<MTOK, blk, 0, stream>>>(of, gf, gob);
    gemm_f16_k<3><<<gridP, blk, 0, stream>>>(gob, WoT + l * HH, yf, nullptr,
                                             curx, MTOK, HDIM, HDIM, SEQ);
    layernorm_k<<<MTOK, blk, 0, stream>>>(yf, ynb, ln_g, ln_b);
    gemm_f16_k<4><<<gridF, blk, 0, stream>>>(ynb, W1T + (size_t)l * FFDIM * HDIM,
                                             hb, b1 + (size_t)l * FFDIM, nullptr,
                                             MTOK, FFDIM, HDIM, SEQ);
    float* outx = (l == LAYERS - 1) ? (float*)d_out : xf;
    gemm_f16_k<5><<<gridP, blk, 0, stream>>>(hb, W2T + (size_t)l * HDIM * FFDIM,
                                             outx, b2 + (size_t)l * HDIM, yf,
                                             MTOK, HDIM, FFDIM, SEQ);
    curx = outx;
  }
}
